// DeepSeekDecoderBlock_47373489275242
// MI455X (gfx1250) — compile-verified, matrix operand order checked by reading, not run on hardware
//
#include <hip/hip_runtime.h>
#include <stdint.h>

// ---------------------------------------------------------------------------
// Types
// ---------------------------------------------------------------------------
typedef __bf16 bf16;
typedef bf16  v16bf __attribute__((ext_vector_type(16)));
typedef float v8f   __attribute__((ext_vector_type(8)));
typedef unsigned short u16;

// ---------------------------------------------------------------------------
// Problem constants (match reference)
// ---------------------------------------------------------------------------
#define B_     2
#define S_     2048
#define DIM_   2048
#define NH_    16
#define NOPE_  128
#define ROPE_  64
#define QKD_   192          // NOPE + ROPE
#define VH_    128
#define KVR_   512
#define NE_    8
#define MINTER_ 1408
#define SHI_   2816
#define T_     (B_*S_)      // 4096 tokens
#define EPS_   1e-6f

// ---------------------------------------------------------------------------
// CDNA5 TDM availability (5-arg on ROCm clang-22, 6-arg on therock clang-23)
// ---------------------------------------------------------------------------
#if defined(__gfx1250__) && __has_builtin(__builtin_amdgcn_tensor_load_to_lds) && \
    __has_builtin(__builtin_amdgcn_s_wait_tensorcnt)
#define CDNA5_TDM 1
#else
#define CDNA5_TDM 0
#endif
#if __has_include(<hip/amd_detail/amd_gfx1250_TDM.h>)
#define CDNA5_TDM6 1
#else
#define CDNA5_TDM6 0
#endif

#if CDNA5_TDM
typedef unsigned int u32x4 __attribute__((ext_vector_type(4)));
typedef int          i32x8 __attribute__((ext_vector_type(8)));
typedef int          i32x4 __attribute__((ext_vector_type(4)));

// Issue one 2D bf16 tile load Global->LDS through the Tensor Data Mover.
// Tile: tile_rows x tile_cols elements (2B each), row stride = ld elements.
// tensor dims are the REMAINING extent from the tile origin so hardware OOB
// handling zero-fills the N/K edges for us. LDS destination uses our padded
// layout: 32-elem (64B = 16 DWORD) rows + 8-elem (16B = 4 DWORD) pad, i.e.
// pad_interval code 3, pad_amount code 3.
__device__ __forceinline__ void tdm_load_2d_bf16(
    const u16* __restrict__ gsrc, unsigned lds_byte_addr,
    long row_stride_elems, int rem_cols, int rem_rows,
    int tile_cols, int tile_rows)
{
  const unsigned long long ga = (unsigned long long)(uintptr_t)gsrc;
  u32x4 g0;
  g0[0] = 1u;                                              // count=1, no gather
  g0[1] = lds_byte_addr;                                   // lds_addr
  g0[2] = (unsigned)ga;                                    // global_addr[31:0]
  g0[3] = ((unsigned)(ga >> 32) & 0x01FFFFFFu) | (2u << 30); // addr[56:32], type=2
  const unsigned long long st = (unsigned long long)row_stride_elems;
  i32x8 g1;
  g1[0] = (1 << 16) | (1 << 20) | (3 << 22) | (3 << 25);   // data_size=2B, pad en, int=16dw, amt=4dw
  g1[1] = (int)(((unsigned)rem_cols & 0xFFFFu) << 16);                 // tensor_dim0[15:0]
  g1[2] = (int)(((unsigned)rem_cols >> 16) | (((unsigned)rem_rows & 0xFFFFu) << 16));
  g1[3] = (int)(((unsigned)rem_rows >> 16) | ((unsigned)tile_cols << 16)); // tile_dim0
  g1[4] = tile_rows & 0xFFFF;                              // tile_dim1 (tile_dim2=0)
  g1[5] = (int)(unsigned)st;                               // dim0_stride[31:0]
  g1[6] = (int)(unsigned)((st >> 32) & 0xFFFFu);           // dim0_stride[47:32]
  g1[7] = 0;
  i32x4 g2; g2[0] = g2[1] = g2[2] = g2[3] = 0;
  i32x4 g3; g3[0] = g3[1] = g3[2] = g3[3] = 0;
#if CDNA5_TDM6
  i32x8 g4; g4[0]=g4[1]=g4[2]=g4[3]=g4[4]=g4[5]=g4[6]=g4[7]=0;
  __builtin_amdgcn_tensor_load_to_lds(g0, g1, g2, g3, g4, 0);
#else
  __builtin_amdgcn_tensor_load_to_lds(g0, g1, g2, g3, 0);
#endif
}
#endif // CDNA5_TDM

// ---------------------------------------------------------------------------
// bf16 <-> f32 helpers (round-to-nearest-even)
// ---------------------------------------------------------------------------
__device__ __forceinline__ u16 f2bf(float f) {
  union { float f; uint32_t u; } c; c.f = f;
  uint32_t u = c.u;
  u += 0x7FFFu + ((u >> 16) & 1u);
  return (u16)(u >> 16);
}
__device__ __forceinline__ float bf2f(u16 h) {
  union { uint32_t u; float f; } c; c.u = ((uint32_t)h) << 16;
  return c.f;
}

// ---------------------------------------------------------------------------
// Manual staging helpers (fallback path + transposed stage for NN B operand)
// LDS row pitch fixed at 40 bf16 (32 data + 8 pad).
// ---------------------------------------------------------------------------
__device__ __forceinline__ void stage_rowmajor_tile(
    const u16* __restrict__ src00, long ld, int rows_valid,
    bf16* __restrict__ dst, int tid)
{
  const int r = tid >> 1;            // 0..127
  const int c = (tid & 1) * 16;      // 0 or 16
  if (r < rows_valid) {
    const u16* s = src00 + (long)r * ld + c;
    uint4 p0 = ((const uint4*)s)[0];
    uint4 p1 = ((const uint4*)s)[1];
    *(uint4*)&dst[r * 40 + c]     = p0;
    *(uint4*)&dst[r * 40 + c + 8] = p1;
    __builtin_prefetch(s + 32, 0, 3);
  } else {
    uint4 zz; zz.x = zz.y = zz.z = zz.w = 0u;
    *(uint4*)&dst[r * 40 + c]     = zz;
    *(uint4*)&dst[r * 40 + c + 8] = zz;
  }
}
__device__ __forceinline__ void stage_colmajor_tile(
    const u16* __restrict__ Bp, long ldb, int k0, int n0, int N,
    bf16* __restrict__ dst, int tid)
{
  const int kk  = tid >> 3;          // 0..31
  const int nof = (tid & 7) * 16;    // 0..112
  const u16* src = Bp + (long)(k0 + kk) * ldb + (n0 + nof);
  bf16 tmp[16];
  uint4 p0 = ((const uint4*)src)[0];
  uint4 p1 = ((const uint4*)src)[1];
  *(uint4*)&tmp[0] = p0;
  *(uint4*)&tmp[8] = p1;
#pragma unroll
  for (int i = 0; i < 16; ++i)
    dst[(nof + i) * 40 + kk] = (n0 + nof + i < N) ? tmp[i] : (bf16)0.0f;
}

// ---------------------------------------------------------------------------
// Generic bf16 WMMA GEMM: C[M,N] = A[M,K] * op(B)
//   TRANS_B = true : B is weight-style [N,K] row-major (C = A @ B^T)
//   TRANS_B = false: B is [K,N] row-major (C = A @ B)
// 128x128 tile, K-step 32, 256 threads = 8 waves, each wave: 4x2 WMMA tiles.
// Double-buffered LDS; tile staging via TDM (tensor_load_to_lds +
// s_wait_tensorcnt) when available, manual global->LDS copies otherwise.
// Batched via blockIdx.z with element strides sA/sB/sC.
// ---------------------------------------------------------------------------
template<bool TRANS_B, bool OUT_BF16>
__global__ __launch_bounds__(256) void k_gemm(
    const u16* __restrict__ A, long sA, int lda,
    const u16* __restrict__ Bm, long sB, int ldb,
    void* __restrict__ Cv, long sC, int ldc,
    int M, int N, int K)
{
  constexpr int TM = 128, TN = 128, TK = 32, LP = 40;
  __shared__ bf16 As[2][TM * LP];
  __shared__ bf16 Bs[2][TN * LP];

  const int z  = blockIdx.z;
  const u16* Ap = A  + (long)z * sA;
  const u16* Bp = Bm + (long)z * sB;

  const int n0   = blockIdx.x * TN;
  const int m0   = blockIdx.y * TM;
  const int tid  = threadIdx.x;
  const int lane = tid & 31;
  const int wave = tid >> 5;     // 0..7
  const int wm   = wave & 1;     // 2 waves along M  (2*64 = 128)
  const int wn   = wave >> 1;    // 4 waves along N  (4*32 = 128)
  const int half = lane >> 4;    // lane group 0/1
  const int lrow = lane & 15;

  v8f acc[4][2];
#pragma unroll
  for (int i = 0; i < 4; ++i)
#pragma unroll
    for (int j = 0; j < 2; ++j)
#pragma unroll
      for (int r = 0; r < 8; ++r) acc[i][j][r] = 0.0f;

  const int nIter = K / TK;

  // ---- prologue: stage K-tile 0 into buffer 0 ----
#if CDNA5_TDM
  constexpr int PEND = TRANS_B ? 2 : 1;   // TDM ops issued per stage
  if (tid == 0) {
    tdm_load_2d_bf16(Ap + (long)m0 * lda,
                     (unsigned)(uintptr_t)&As[0][0],
                     lda, K, M - m0, TK, TM);
    if (TRANS_B)
      tdm_load_2d_bf16(Bp + (long)n0 * ldb,
                       (unsigned)(uintptr_t)&Bs[0][0],
                       ldb, K, N - n0, TK, TN);
  }
  if (!TRANS_B) stage_colmajor_tile(Bp, ldb, 0, n0, N, &Bs[0][0], tid);
#else
  stage_rowmajor_tile(Ap + (long)m0 * lda, lda, TM, &As[0][0], tid);
  if (TRANS_B) {
    const int nv = N - n0;
    stage_rowmajor_tile(Bp + (long)n0 * ldb, ldb, nv < TN ? nv : TN, &Bs[0][0], tid);
  } else {
    stage_colmajor_tile(Bp, ldb, 0, n0, N, &Bs[0][0], tid);
  }
#endif

  for (int it = 0; it < nIter; ++it) {
    const int cur = it & 1;
    const int nxt = cur ^ 1;
    const int kn  = (it + 1) * TK;
    const bool more = (it + 1) < nIter;

    // ---- stage next K-tile into the other buffer while we compute ----
#if CDNA5_TDM
    if (more && tid == 0) {
      tdm_load_2d_bf16(Ap + (long)m0 * lda + kn,
                       (unsigned)(uintptr_t)&As[nxt][0],
                       lda, K - kn, M - m0, TK, TM);
      if (TRANS_B)
        tdm_load_2d_bf16(Bp + (long)n0 * ldb + kn,
                         (unsigned)(uintptr_t)&Bs[nxt][0],
                         ldb, K - kn, N - n0, TK, TN);
    }
    if (!TRANS_B && more) stage_colmajor_tile(Bp, ldb, kn, n0, N, &Bs[nxt][0], tid);
    // retire the current tile's DMA (in-order per wave; only the next stage's
    // PEND ops may remain outstanding). Non-issuing waves have TENSORcnt==0.
    if (more) __builtin_amdgcn_s_wait_tensorcnt(PEND);
    else      __builtin_amdgcn_s_wait_tensorcnt(0);
#else
    if (more) {
      stage_rowmajor_tile(Ap + (long)m0 * lda + kn, lda, TM, &As[nxt][0], tid);
      if (TRANS_B) {
        const int nv = N - n0;
        stage_rowmajor_tile(Bp + (long)n0 * ldb + kn, ldb, nv < TN ? nv : TN,
                            &Bs[nxt][0], tid);
      } else {
        stage_colmajor_tile(Bp, ldb, kn, n0, N, &Bs[nxt][0], tid);
      }
    }
#endif
    __syncthreads();

    // ---- build fragments per the 16-bit A/B VGPR layout ----
    v16bf af[4], bfr[2];
#pragma unroll
    for (int mt = 0; mt < 4; ++mt) {
      const bf16* ap = &As[cur][(wm * 64 + mt * 16 + lrow) * LP];
#pragma unroll
      for (int j = 0; j < 8; ++j) {
        const int kb = ((j < 4) ? (2 * j) : (16 + 2 * (j - 4))) + half * 8;
        af[mt][2 * j]     = ap[kb];
        af[mt][2 * j + 1] = ap[kb + 1];
      }
    }
#pragma unroll
    for (int nt = 0; nt < 2; ++nt) {
      const bf16* bp = &Bs[cur][(wn * 32 + nt * 16 + lrow) * LP];
#pragma unroll
      for (int j = 0; j < 8; ++j) {
        const int kb = ((j < 4) ? (2 * j) : (16 + 2 * (j - 4))) + half * 8;
        bfr[nt][2 * j]     = bp[kb];
        bfr[nt][2 * j + 1] = bp[kb + 1];
      }
    }

#pragma unroll
    for (int mt = 0; mt < 4; ++mt)
#pragma unroll
      for (int nt = 0; nt < 2; ++nt)
        acc[mt][nt] = __builtin_amdgcn_wmma_f32_16x16x32_bf16(
            false, af[mt], false, bfr[nt], (short)0, acc[mt][nt], false, false);

    __syncthreads();
  }

  // ---- epilogue: C 16x16 layout (VGPR r: lanes 0-15 -> M=r, lanes 16-31 -> M=r+8)
  const long cz = (long)z * sC;
#pragma unroll
  for (int mt = 0; mt < 4; ++mt) {
#pragma unroll
    for (int nt = 0; nt < 2; ++nt) {
      const int col = n0 + wn * 32 + nt * 16 + lrow;
      if (col < N) {
#pragma unroll
        for (int r = 0; r < 8; ++r) {
          const int row = m0 + wm * 64 + mt * 16 + half * 8 + r;
          const float v = acc[mt][nt][r];
          if (OUT_BF16) ((u16*)Cv)[cz + (long)row * ldc + col] = f2bf(v);
          else          ((float*)Cv)[cz + (long)row * ldc + col] = v;
        }
      }
    }
  }
}

// ---------------------------------------------------------------------------
// fp32 -> bf16 conversion (grid-stride)
// ---------------------------------------------------------------------------
__global__ void k_f32_to_bf16(const float* __restrict__ s, u16* __restrict__ d, long n) {
  long i  = (long)blockIdx.x * blockDim.x + threadIdx.x;
  long st = (long)gridDim.x * blockDim.x;
  for (; i < n; i += st) d[i] = f2bf(s[i]);
}

// ---------------------------------------------------------------------------
// RMSNorm: one block per row, fp32 in (row stride in_ld), bf16 out
// ---------------------------------------------------------------------------
__global__ __launch_bounds__(256) void k_rmsnorm(
    const float* __restrict__ in, int in_ld,
    const float* __restrict__ w,
    u16* __restrict__ outp, int out_ld, int width)
{
  const int row = blockIdx.x;
  const float* xr = in + (long)row * in_ld;
  __shared__ float sh[8];
  float ss = 0.f;
  for (int i = threadIdx.x; i < width; i += 256) { float v = xr[i]; ss += v * v; }
  for (int o = 16; o; o >>= 1) ss += __shfl_down(ss, o, 32);
  if ((threadIdx.x & 31) == 0) sh[threadIdx.x >> 5] = ss;
  __syncthreads();
  if (threadIdx.x == 0) { float t = 0.f; for (int k = 0; k < 8; ++k) t += sh[k]; sh[0] = t; }
  __syncthreads();
  const float inv = rsqrtf(sh[0] / (float)width + EPS_);
  u16* orow = outp + (long)row * out_ld;
  for (int i = threadIdx.x; i < width; i += 256) orow[i] = f2bf(xr[i] * inv * w[i]);
}

// ---------------------------------------------------------------------------
// Build per-head Q (copy NOPE, rope the ROPE part): qf [T, NH*192] fp32 ->
// Qh [B*NH, S, 192] bf16.  grid = T*NH blocks, 192 threads.
// ---------------------------------------------------------------------------
__global__ void k_build_q(const float* __restrict__ qf,
                          const float* __restrict__ cosT,
                          const float* __restrict__ sinT,
                          u16* __restrict__ Qh)
{
  const int idx = blockIdx.x;          // ((b*S)+s)*NH + h
  const int h  = idx % NH_;
  const int bs = idx / NH_;
  const int s  = bs % S_;
  const int b  = bs / S_;
  const int d  = threadIdx.x;          // 0..191
  const float* src = qf + (long)bs * (NH_ * QKD_) + h * QKD_;
  float v;
  if (d < NOPE_) {
    v = src[d];
  } else {
    const int p    = d - NOPE_;
    const int pair = p >> 1;
    const float c  = cosT[s * (ROPE_ / 2) + pair];
    const float sn = sinT[s * (ROPE_ / 2) + pair];
    const float x1 = src[NOPE_ + 2 * pair];
    const float x2 = src[NOPE_ + 2 * pair + 1];
    v = (p & 1) ? (x1 * sn + x2 * c) : (x1 * c - x2 * sn);
  }
  Qh[(((long)(b * NH_ + h)) * S_ + s) * QKD_ + d] = f2bf(v);
}

// ---------------------------------------------------------------------------
// Build per-head K (k_nope + roped broadcast k_pe) and V.
// kvbf [T, NH*256] fp32; kvf [T, 576] fp32 (pe at 512..575).
// Kh [B*NH, S, 192], Vh [B*NH, S, 128].  grid = T*NH, 192 threads.
// ---------------------------------------------------------------------------
__global__ void k_build_kv(const float* __restrict__ kvbf,
                           const float* __restrict__ kvf,
                           const float* __restrict__ cosT,
                           const float* __restrict__ sinT,
                           u16* __restrict__ Kh, u16* __restrict__ Vh)
{
  const int idx = blockIdx.x;
  const int h  = idx % NH_;
  const int bs = idx / NH_;
  const int s  = bs % S_;
  const int b  = bs / S_;
  const int d  = threadIdx.x;          // 0..191
  const float* src = kvbf + (long)bs * (NH_ * (NOPE_ + VH_)) + h * (NOPE_ + VH_);
  float v;
  if (d < NOPE_) {
    v = src[d];
  } else {
    const int p    = d - NOPE_;
    const int pair = p >> 1;
    const float* pe = kvf + (long)bs * (KVR_ + ROPE_) + KVR_;
    const float c  = cosT[s * (ROPE_ / 2) + pair];
    const float sn = sinT[s * (ROPE_ / 2) + pair];
    const float x1 = pe[2 * pair];
    const float x2 = pe[2 * pair + 1];
    v = (p & 1) ? (x1 * sn + x2 * c) : (x1 * c - x2 * sn);
  }
  const long bh = (long)(b * NH_ + h);
  Kh[(bh * S_ + s) * QKD_ + d] = f2bf(v);
  if (d < VH_) Vh[(bh * S_ + s) * VH_ + d] = f2bf(src[NOPE_ + d]);
}

// ---------------------------------------------------------------------------
// Causal softmax in-place on bf16 scores; one block per row.
// ---------------------------------------------------------------------------
__global__ __launch_bounds__(256) void k_softmax_causal(u16* __restrict__ sc, float scale)
{
  const long row = blockIdx.x;             // 0 .. B*NH*S-1
  const int  i   = (int)(row % S_);        // causal limit
  u16* p = sc + row * (long)S_;
  __shared__ float sh[8];
  const int tid = threadIdx.x;

  float mx = -3.0e38f;
  for (int j = tid; j <= i; j += 256) mx = fmaxf(mx, bf2f(p[j]) * scale);
  for (int o = 16; o; o >>= 1) mx = fmaxf(mx, __shfl_down(mx, o, 32));
  if ((tid & 31) == 0) sh[tid >> 5] = mx;
  __syncthreads();
  if (tid == 0) { float m = sh[0]; for (int k = 1; k < 8; ++k) m = fmaxf(m, sh[k]); sh[0] = m; }
  __syncthreads();
  mx = sh[0];
  __syncthreads();

  float sum = 0.f;
  for (int j = tid; j <= i; j += 256) sum += __expf(bf2f(p[j]) * scale - mx);
  for (int o = 16; o; o >>= 1) sum += __shfl_down(sum, o, 32);
  if ((tid & 31) == 0) sh[tid >> 5] = sum;
  __syncthreads();
  if (tid == 0) { float t = 0.f; for (int k = 0; k < 8; ++k) t += sh[k]; sh[0] = t; }
  __syncthreads();
  const float inv = 1.f / sh[0];

  for (int j = tid; j < S_; j += 256) {
    const float o = (j <= i) ? __expf(bf2f(p[j]) * scale - mx) * inv : 0.f;
    p[j] = f2bf(o);
  }
}

// ---------------------------------------------------------------------------
// Merge heads: Oh [B*NH, S, 128] fp32 -> attnb [T, NH*128] bf16
// ---------------------------------------------------------------------------
__global__ void k_merge_attn(const float* __restrict__ Oh, u16* __restrict__ attnb)
{
  const int ts = blockIdx.x;       // token index
  const int b  = ts / S_;
  const int s  = ts % S_;
  for (int c = threadIdx.x; c < NH_ * VH_; c += blockDim.x) {
    const int h = c / VH_, d = c % VH_;
    const float v = Oh[(((long)(b * NH_ + h)) * S_ + s) * VH_ + d];
    attnb[(long)ts * (NH_ * VH_) + c] = f2bf(v);
  }
}

// ---------------------------------------------------------------------------
// Elementwise helpers
// ---------------------------------------------------------------------------
__global__ void k_add_inplace(float* __restrict__ a, const float* __restrict__ b, long n) {
  long i = (long)blockIdx.x * blockDim.x + threadIdx.x;
  long st = (long)gridDim.x * blockDim.x;
  for (; i < n; i += st) a[i] += b[i];
}
__global__ void k_silu_mul(const float* __restrict__ a1, const float* __restrict__ a3,
                           u16* __restrict__ h1, long n) {
  long i = (long)blockIdx.x * blockDim.x + threadIdx.x;
  long st = (long)gridDim.x * blockDim.x;
  for (; i < n; i += st) {
    const float g = a1[i];
    h1[i] = f2bf((g / (1.f + __expf(-g))) * a3[i]);
  }
}
__global__ void k_moe_accum(float* __restrict__ facc, const float* __restrict__ eo,
                            const float* __restrict__ gates, int e, long n) {
  long i = (long)blockIdx.x * blockDim.x + threadIdx.x;
  long st = (long)gridDim.x * blockDim.x;
  for (; i < n; i += st) {
    const long t = i / DIM_;
    facc[i] += gates[t * NE_ + e] * eo[i];
  }
}
__global__ void k_final(float* __restrict__ o, const float* __restrict__ h,
                        const float* __restrict__ f, long n) {
  long i = (long)blockIdx.x * blockDim.x + threadIdx.x;
  long st = (long)gridDim.x * blockDim.x;
  for (; i < n; i += st) o[i] = h[i] + f[i];
}

// ---------------------------------------------------------------------------
// Router: softmax over 8 experts, top-2, renormalized gates.
// ---------------------------------------------------------------------------
__global__ void k_router_topk(const float* __restrict__ logits, float* __restrict__ gates)
{
  const int t = blockIdx.x * blockDim.x + threadIdx.x;
  if (t >= T_) return;
  float l[NE_];
  float mx = -3.0e38f;
  for (int e = 0; e < NE_; ++e) { l[e] = logits[t * NE_ + e]; mx = fmaxf(mx, l[e]); }
  float s = 0.f;
  for (int e = 0; e < NE_; ++e) { l[e] = __expf(l[e] - mx); s += l[e]; }
  for (int e = 0; e < NE_; ++e) l[e] /= s;
  int i1 = 0;
  for (int e = 1; e < NE_; ++e) if (l[e] > l[i1]) i1 = e;
  int i2 = -1;
  for (int e = 0; e < NE_; ++e) if (e != i1 && (i2 < 0 || l[e] > l[i2])) i2 = e;
  const float d = l[i1] + l[i2];
  for (int e = 0; e < NE_; ++e) gates[t * NE_ + e] = 0.f;
  gates[t * NE_ + i1] = l[i1] / d;
  gates[t * NE_ + i2] = l[i2] / d;
}

// ---------------------------------------------------------------------------
// Host-side orchestration
// ---------------------------------------------------------------------------
extern "C" void kernel_launch(void* const* d_in, const int* in_sizes, int n_in,
                              void* d_out, int out_size, void* d_ws, size_t ws_size,
                              hipStream_t stream)
{
  (void)in_sizes; (void)n_in; (void)out_size; (void)ws_size;

  const float* x     = (const float*)d_in[0];
  const float* cosT  = (const float*)d_in[1];
  const float* sinT  = (const float*)d_in[2];
  const float* anw   = (const float*)d_in[3];
  const float* wq    = (const float*)d_in[4];
  const float* wkv_a = (const float*)d_in[5];
  const float* kvnw  = (const float*)d_in[6];
  const float* wkv_b = (const float*)d_in[7];
  const float* wo    = (const float*)d_in[8];
  const float* ffnw  = (const float*)d_in[9];
  const float* rw    = (const float*)d_in[10];
  const float* e_w1  = (const float*)d_in[11];
  const float* e_w3  = (const float*)d_in[12];
  const float* e_w2  = (const float*)d_in[13];
  const float* s_w1  = (const float*)d_in[14];
  const float* s_w3  = (const float*)d_in[15];
  const float* s_w2  = (const float*)d_in[16];
  float* out = (float*)d_out;

  char* base = (char*)d_ws;
  size_t off = 0;
  auto alloc = [&](size_t bytes) -> void* {
    void* p = base + off;
    off = (off + bytes + 255) & ~(size_t)255;
    return p;
  };
  auto cvt = [&](const float* src, size_t n) -> u16* {
    u16* dst = (u16*)alloc(n * sizeof(u16));
    k_f32_to_bf16<<<2048, 256, 0, stream>>>(src, dst, (long)n);
    return dst;
  };

  // ---- bf16 weight copies (weights are small vs FLOPs; L2 is 192MB) ----
  u16* wq_b   = cvt(wq,    (size_t)NH_ * QKD_ * DIM_);
  u16* wkva_b = cvt(wkv_a, (size_t)(KVR_ + ROPE_) * DIM_);
  u16* wkvb_b = cvt(wkv_b, (size_t)NH_ * (NOPE_ + VH_) * KVR_);
  u16* wo_b   = cvt(wo,    (size_t)DIM_ * NH_ * VH_);
  u16* rw_b   = cvt(rw,    (size_t)NE_ * DIM_);
  u16* ew1_b  = cvt(e_w1,  (size_t)NE_ * MINTER_ * DIM_);
  u16* ew3_b  = cvt(e_w3,  (size_t)NE_ * MINTER_ * DIM_);
  u16* ew2_b  = cvt(e_w2,  (size_t)NE_ * DIM_ * MINTER_);
  u16* sw1_b  = cvt(s_w1,  (size_t)SHI_ * DIM_);
  u16* sw3_b  = cvt(s_w3,  (size_t)SHI_ * DIM_);
  u16* sw2_b  = cvt(s_w2,  (size_t)DIM_ * SHI_);

  // ---- activations ----
  u16*   nx    = (u16*)  alloc((size_t)T_ * DIM_ * 2);
  float* qf    = (float*)alloc((size_t)T_ * NH_ * QKD_ * 4);
  float* kvf   = (float*)alloc((size_t)T_ * (KVR_ + ROPE_) * 4);
  u16*   ckv   = (u16*)  alloc((size_t)T_ * KVR_ * 2);
  float* kvbf  = (float*)alloc((size_t)T_ * NH_ * (NOPE_ + VH_) * 4);
  u16*   Qh    = (u16*)  alloc((size_t)B_ * NH_ * S_ * QKD_ * 2);
  u16*   Kh    = (u16*)  alloc((size_t)B_ * NH_ * S_ * QKD_ * 2);
  u16*   Vh    = (u16*)  alloc((size_t)B_ * NH_ * S_ * VH_ * 2);
  u16*   Pm    = (u16*)  alloc((size_t)B_ * NH_ * S_ * S_ * 2);   // scores -> probs
  float* Oh    = (float*)alloc((size_t)B_ * NH_ * S_ * VH_ * 4);
  u16*   attnb = (u16*)  alloc((size_t)T_ * NH_ * VH_ * 2);
  float* hf    = (float*)alloc((size_t)T_ * DIM_ * 4);
  u16*   nh    = (u16*)  alloc((size_t)T_ * DIM_ * 2);
  float* rlog  = (float*)alloc((size_t)T_ * NE_ * 4);
  float* gates = (float*)alloc((size_t)T_ * NE_ * 4);
  float* a1    = (float*)alloc((size_t)T_ * SHI_ * 4);
  float* a3    = (float*)alloc((size_t)T_ * SHI_ * 4);
  u16*   h1    = (u16*)  alloc((size_t)T_ * SHI_ * 2);
  float* eo    = (float*)alloc((size_t)T_ * DIM_ * 4);
  float* facc  = (float*)alloc((size_t)T_ * DIM_ * 4);

  auto gemm_tn = [&](const u16* A, long sA, int lda, const u16* Bp, long sB, int ldb,
                     void* C, long sC, int ldc, int M, int N, int K, int Z, bool outbf) {
    dim3 g((unsigned)((N + 127) / 128), (unsigned)(M / 128), (unsigned)Z);
    if (outbf) k_gemm<true,  true ><<<g, 256, 0, stream>>>(A, sA, lda, Bp, sB, ldb, C, sC, ldc, M, N, K);
    else       k_gemm<true,  false><<<g, 256, 0, stream>>>(A, sA, lda, Bp, sB, ldb, C, sC, ldc, M, N, K);
  };
  auto gemm_nn = [&](const u16* A, long sA, int lda, const u16* Bp, long sB, int ldb,
                     void* C, long sC, int ldc, int M, int N, int K, int Z) {
    dim3 g((unsigned)((N + 127) / 128), (unsigned)(M / 128), (unsigned)Z);
    k_gemm<false, false><<<g, 256, 0, stream>>>(A, sA, lda, Bp, sB, ldb, C, sC, ldc, M, N, K);
  };

  // ---- attention ----
  k_rmsnorm<<<T_, 256, 0, stream>>>(x, DIM_, anw, nx, DIM_, DIM_);
  gemm_tn(nx, 0, DIM_, wq_b, 0, DIM_, qf, 0, NH_ * QKD_, T_, NH_ * QKD_, DIM_, 1, false);
  gemm_tn(nx, 0, DIM_, wkva_b, 0, DIM_, kvf, 0, KVR_ + ROPE_, T_, KVR_ + ROPE_, DIM_, 1, false);
  k_rmsnorm<<<T_, 256, 0, stream>>>(kvf, KVR_ + ROPE_, kvnw, ckv, KVR_, KVR_);
  gemm_tn(ckv, 0, KVR_, wkvb_b, 0, KVR_, kvbf, 0, NH_ * (NOPE_ + VH_),
          T_, NH_ * (NOPE_ + VH_), KVR_, 1, false);
  k_build_q<<<T_ * NH_, 192, 0, stream>>>(qf, cosT, sinT, Qh);
  k_build_kv<<<T_ * NH_, 192, 0, stream>>>(kvbf, kvf, cosT, sinT, Kh, Vh);

  // scores[z] = Qh[z] @ Kh[z]^T  (bf16 out, batched over B*NH)
  gemm_tn(Qh, (long)S_ * QKD_, QKD_, Kh, (long)S_ * QKD_, QKD_,
          Pm, (long)S_ * S_, S_, S_, S_, QKD_, B_ * NH_, true);
  k_softmax_causal<<<B_ * NH_ * S_, 256, 0, stream>>>(Pm, rsqrtf((float)QKD_));
  // Oh[z] = P[z] @ V[z]
  gemm_nn(Pm, (long)S_ * S_, S_, Vh, (long)S_ * VH_, VH_,
          Oh, (long)S_ * VH_, VH_, S_, VH_, S_, B_ * NH_);
  k_merge_attn<<<T_, 256, 0, stream>>>(Oh, attnb);
  gemm_tn(attnb, 0, NH_ * VH_, wo_b, 0, NH_ * VH_, hf, 0, DIM_, T_, DIM_, NH_ * VH_, 1, false);
  k_add_inplace<<<2048, 256, 0, stream>>>(hf, x, (long)T_ * DIM_);

  // ---- FFN: router + shared + 8 experts (dense-gated, matches reference) ----
  k_rmsnorm<<<T_, 256, 0, stream>>>(hf, DIM_, ffnw, nh, DIM_, DIM_);
  gemm_tn(nh, 0, DIM_, rw_b, 0, DIM_, rlog, 0, NE_, T_, NE_, DIM_, 1, false);
  k_router_topk<<<(T_ + 255) / 256, 256, 0, stream>>>(rlog, gates);

  // shared expert -> facc
  gemm_tn(nh, 0, DIM_, sw1_b, 0, DIM_, a1, 0, SHI_, T_, SHI_, DIM_, 1, false);
  gemm_tn(nh, 0, DIM_, sw3_b, 0, DIM_, a3, 0, SHI_, T_, SHI_, DIM_, 1, false);
  k_silu_mul<<<2048, 256, 0, stream>>>(a1, a3, h1, (long)T_ * SHI_);
  gemm_tn(h1, 0, SHI_, sw2_b, 0, SHI_, facc, 0, DIM_, T_, DIM_, SHI_, 1, false);

  // routed experts, gate-weighted accumulation into facc
  for (int e = 0; e < NE_; ++e) {
    const u16* w1e = ew1_b + (size_t)e * MINTER_ * DIM_;
    const u16* w3e = ew3_b + (size_t)e * MINTER_ * DIM_;
    const u16* w2e = ew2_b + (size_t)e * DIM_ * MINTER_;
    gemm_tn(nh, 0, DIM_, w1e, 0, DIM_, a1, 0, MINTER_, T_, MINTER_, DIM_, 1, false);
    gemm_tn(nh, 0, DIM_, w3e, 0, DIM_, a3, 0, MINTER_, T_, MINTER_, DIM_, 1, false);
    k_silu_mul<<<2048, 256, 0, stream>>>(a1, a3, h1, (long)T_ * MINTER_);
    gemm_tn(h1, 0, MINTER_, w2e, 0, MINTER_, eo, 0, DIM_, T_, DIM_, MINTER_, 1, false);
    k_moe_accum<<<2048, 256, 0, stream>>>(facc, eo, gates, e, (long)T_ * DIM_);
  }

  k_final<<<2048, 256, 0, stream>>>(out, hf, facc, (long)T_ * DIM_);
}
